// HgnnEncoder_35038343201423
// MI455X (gfx1250) — matrix operand order repaired and checked by reading, not compiled, not run on hardware
//
#include <hip/hip_runtime.h>

// ---------------- problem constants (match reference) ----------------
#define N_NODES  100000
#define N_EDGES  20000
#define NNZ_CNT  1600000
#define IN_C     256
#define OUT_CC   256
#define HID_C    128
#define BN_EPS   1e-5f

typedef __bf16 bf16_t;
typedef __attribute__((ext_vector_type(16))) __bf16 v16bf;
typedef __attribute__((ext_vector_type(8)))  __bf16 v8bf;
typedef __attribute__((ext_vector_type(8)))  float  v8f;

// ---------------- utility kernels ----------------
__global__ void zero_f32_k(float* __restrict__ p, long long n) {
  long long i = (long long)blockIdx.x * blockDim.x + threadIdx.x;
  long long stride = (long long)gridDim.x * blockDim.x;
  for (; i < n; i += stride) p[i] = 0.f;
}

__global__ void f32_to_bf16_k(const float* __restrict__ in, bf16_t* __restrict__ out, long long n) {
  long long i = (long long)blockIdx.x * blockDim.x + threadIdx.x;
  long long stride = (long long)gridDim.x * blockDim.x;
  for (; i < n; i += stride) out[i] = (bf16_t)in[i];
}

__global__ void degree_k(const int* __restrict__ node_idx, const int* __restrict__ hedge_idx,
                         float* __restrict__ deg_n, float* __restrict__ deg_e) {
  int i = blockIdx.x * blockDim.x + threadIdx.x;
  if (i < NNZ_CNT) {
    atomicAdd(&deg_n[node_idx[i]], 1.f);
    atomicAdd(&deg_e[hedge_idx[i]], 1.f);
  }
}

__global__ void invert_k(float* __restrict__ p, int n) {
  int i = blockIdx.x * blockDim.x + threadIdx.x;
  if (i < n) { float d = p[i]; p[i] = (d > 0.f) ? (1.f / d) : 0.f; }
}

// ---------------- weight packing into WMMA B-fragment layout ----------------
// B (32x16 bf16 per k-tile): element j (0..15) of lane L holds
//   B[K = kt*32 + (j/8)*16 + (L/16)*8 + (j%8)][N = ntile*16 + (L%16)]
// packed index p = ((ntile*KT + kt)*32 + lane)*16 + j  -> 32B contiguous per lane.
__global__ void pack_b_k(const float* __restrict__ w, bf16_t* __restrict__ bp, int K, int C) {
  int p = blockIdx.x * blockDim.x + threadIdx.x;
  if (p >= K * C) return;
  int j    = p & 15;
  int lane = (p >> 4) & 31;
  int rest = p >> 9;
  int KT   = K >> 5;
  int kt    = rest % KT;
  int ntile = rest / KT;
  int kk = kt * 32 + (j >> 3) * 16 + (lane >> 4) * 8 + (j & 7);
  int nn = ntile * 16 + (lane & 15);
  bp[p] = (bf16_t)w[(size_t)kk * C + nn];
}

// ---------------- WMMA GEMM: C[N x COUT] = A[N x K] * W[K x COUT] ----------------
// 256 threads = 8 waves per block; packed B staged once into LDS and shared by
// all 8 waves; one wave per 16-row M-tile, all COUT/16 N-tiles accumulated per
// wave (A reuse across NT WMMAs, B reuse across the whole block via LDS).
template<int K, int COUT>
__global__ void __launch_bounds__(256)
wmma_gemm_k(const bf16_t* __restrict__ A, const bf16_t* __restrict__ Bp,
            float* __restrict__ C) {
  constexpr int NT = COUT / 16;
  constexpr int KT = K / 32;
  __shared__ bf16_t sB[K * COUT];               // 32-64 KB of the 320 KB WGP LDS

  // cooperative fill (all threads participate, before any wave exits)
  constexpr int NCHUNK = (K * COUT) / 8;        // 16B chunks
  for (int idx = threadIdx.x; idx < NCHUNK; idx += 256)
    ((int4*)sB)[idx] = ((const int4*)Bp)[idx];
  __syncthreads();

  const int lane  = threadIdx.x & 31;
  const int mtile = blockIdx.x * 8 + (threadIdx.x >> 5);
  if (mtile * 16 >= N_NODES) return;            // wave-uniform: EXEC stays all-ones
  const int half = lane >> 4;
  const int mrow = mtile * 16 + (lane & 15);

  v8f acc[NT];
#pragma unroll
  for (int t = 0; t < NT; ++t) acc[t] = (v8f){0.f,0.f,0.f,0.f,0.f,0.f,0.f,0.f};

  const bf16_t* arow = A + (size_t)mrow * K + half * 8;
#pragma unroll
  for (int kt = 0; kt < KT; ++kt) {
    // A-fragment per ISA layout: j=0..7  -> K = kt*32 + half*8 + j      (16B chunk)
    //                            j=8..15 -> K = kt*32 + 16 + half*8 + (j-8) (16B chunk)
    v8bf alo = *(const v8bf*)(arow + kt * 32);
    v8bf ahi = *(const v8bf*)(arow + kt * 32 + 16);
    v16bf afrag = __builtin_shufflevector(alo, ahi, 0,1,2,3,4,5,6,7,8,9,10,11,12,13,14,15);
#pragma unroll
    for (int t = 0; t < NT; ++t) {
      v16bf bfrag = *(const v16bf*)(sB + (((size_t)t * KT + kt) * 32 + lane) * 16);
      acc[t] = __builtin_amdgcn_wmma_f32_16x16x32_bf16(
          /*neg_a=*/false, afrag, /*neg_b=*/false, bfrag,
          /*c_mod=*/(short)0, acc[t], /*reuse_a=*/false, /*reuse_b=*/false);
    }
  }
  // D layout: VGPR r -> M = half*8 + r, N = lane&15
#pragma unroll
  for (int t = 0; t < NT; ++t) {
    float* crow = C + (size_t)(mtile * 16 + half * 8) * COUT + t * 16 + (lane & 15);
#pragma unroll
    for (int r = 0; r < 8; ++r) crow[(size_t)r * COUT] = acc[t][r];
  }
}

// ---------------- scatter: node features -> hyperedge accumulation ----------------
template<int C>
__global__ void scatter_to_edges_k(const float* __restrict__ hw,
                                   const int* __restrict__ node_idx,
                                   const int* __restrict__ hedge_idx,
                                   float* __restrict__ ebuf) {
  constexpr int CG = C / 4;
  long long tid = (long long)blockIdx.x * blockDim.x + threadIdx.x;
  if (tid >= (long long)NNZ_CNT * CG) return;
  int cg = (int)(tid % CG);
  int i  = (int)(tid / CG);
  int nd = node_idx[i], he = hedge_idx[i];
  const float4 v = *(const float4*)(hw + (size_t)nd * C + cg * 4);
  float* dst = ebuf + (size_t)he * C + cg * 4;
  atomicAdd(dst + 0, v.x); atomicAdd(dst + 1, v.y);
  atomicAdd(dst + 2, v.z); atomicAdd(dst + 3, v.w);
}

__global__ void scale_rows_k(float* __restrict__ buf, const float* __restrict__ inv,
                             int C, long long total) {
  long long i = (long long)blockIdx.x * blockDim.x + threadIdx.x;
  long long stride = (long long)gridDim.x * blockDim.x;
  for (; i < total; i += stride) buf[i] *= inv[i / C];
}

// ---------------- scatter: hyperedge features -> node aggregation ----------------
template<int C>
__global__ void scatter_to_nodes_k(const float* __restrict__ ebuf,
                                   const int* __restrict__ node_idx,
                                   const int* __restrict__ hedge_idx,
                                   float* __restrict__ agg) {
  constexpr int CG = C / 4;
  long long tid = (long long)blockIdx.x * blockDim.x + threadIdx.x;
  if (tid >= (long long)NNZ_CNT * CG) return;
  int cg = (int)(tid % CG);
  int i  = (int)(tid / CG);
  int nd = node_idx[i], he = hedge_idx[i];
  const float4 v = *(const float4*)(ebuf + (size_t)he * C + cg * 4);
  float* dst = agg + (size_t)nd * C + cg * 4;
  atomicAdd(dst + 0, v.x); atomicAdd(dst + 1, v.y);
  atomicAdd(dst + 2, v.z); atomicAdd(dst + 3, v.w);
}

// ---------------- post-aggregation: Dinv*agg + bias, ReLU, collect BN stats ----------------
// blockDim.x == C; each block handles a strided set of rows; one atomic per channel/block.
template<int C>
__global__ void post_agg_stats_k(const float* __restrict__ agg, const float* __restrict__ Dinv,
                                 const float* __restrict__ bias, float* __restrict__ hout,
                                 float* __restrict__ sum, float* __restrict__ sumsq) {
  int c = threadIdx.x;
  float bc = bias[c];
  float s = 0.f, s2 = 0.f;
  for (int n = blockIdx.x; n < N_NODES; n += gridDim.x) {
    float t = agg[(size_t)n * C + c] * Dinv[n] + bc;
    t = fmaxf(t, 0.f);
    hout[(size_t)n * C + c] = t;
    s += t; s2 += t * t;
  }
  atomicAdd(&sum[c], s);
  atomicAdd(&sumsq[c], s2);
}

__global__ void bn_finalize_k(float* __restrict__ sum, float* __restrict__ sumsq,
                              const float* __restrict__ g, const float* __restrict__ bt, int C) {
  int c = threadIdx.x;
  if (c < C) {
    float mu   = sum[c] * (1.f / (float)N_NODES);
    float var  = sumsq[c] * (1.f / (float)N_NODES) - mu * mu;
    float rstd = rsqrtf(var + BN_EPS);
    float sc   = rstd * g[c];
    sum[c]   = sc;               // scale
    sumsq[c] = bt[c] - mu * sc;  // shift
  }
}

template<int C>
__global__ void bn_apply_bf16_k(const float* __restrict__ h, const float* __restrict__ scale,
                                const float* __restrict__ shift, bf16_t* __restrict__ out) {
  long long i = (long long)blockIdx.x * blockDim.x + threadIdx.x;
  long long stride = (long long)gridDim.x * blockDim.x;
  const long long total = (long long)N_NODES * C;
  for (; i < total; i += stride) {
    int c = (int)(i % C);
    out[i] = (bf16_t)(h[i] * scale[c] + shift[c]);
  }
}

// ---------------- final layer epilogue (no BN): relu(agg*Dinv + b) -> f32 out ----------------
__global__ void final_out_k(const float* __restrict__ agg, const float* __restrict__ Dinv,
                            const float* __restrict__ bias, float* __restrict__ out) {
  long long i = (long long)blockIdx.x * blockDim.x + threadIdx.x;
  long long stride = (long long)gridDim.x * blockDim.x;
  const long long total = (long long)N_NODES * OUT_CC;
  for (; i < total; i += stride) {
    int c = (int)(i % OUT_CC);
    long long n = i / OUT_CC;
    out[i] = fmaxf(agg[i] * Dinv[n] + bias[c], 0.f);
  }
}

// ---------------- host orchestration ----------------
extern "C" void kernel_launch(void* const* d_in, const int* in_sizes, int n_in,
                              void* d_out, int out_size, void* d_ws, size_t ws_size,
                              hipStream_t stream) {
  (void)in_sizes; (void)n_in; (void)out_size; (void)ws_size;
  const float* Xin[3] = {(const float*)d_in[0], (const float*)d_in[1], (const float*)d_in[2]};
  const int* node_idx  = (const int*)d_in[3];
  const int* hedge_idx = node_idx + NNZ_CNT;

  // workspace carve-out (256B aligned)
  char* ws = (char*)d_ws;
  size_t off = 0;
  auto carve = [&](size_t bytes) -> void* {
    void* p = ws + off;
    off = (off + bytes + 255) & ~(size_t)255;
    return p;
  };
  float*  Dinv = (float*)carve((size_t)N_NODES * 4);
  float*  Binv = (float*)carve((size_t)N_EDGES * 4);
  float*  sums = (float*)carve((size_t)OUT_CC * 4);   // -> BN scale
  float*  sqs  = (float*)carve((size_t)OUT_CC * 4);   // -> BN shift (adjacent to sums)
  bf16_t* Wp   = (bf16_t*)carve((size_t)IN_C * OUT_CC * 2);
  bf16_t* Hb   = (bf16_t*)carve((size_t)N_NODES * IN_C * 2);   // bf16 activations
  float*  HW   = (float*)carve((size_t)N_NODES * OUT_CC * 4);  // GEMM out, reused as node agg
  float*  Eb   = (float*)carve((size_t)N_EDGES * OUT_CC * 4);  // hyperedge accum
  float*  Hf   = (float*)carve((size_t)N_NODES * HID_C * 4);   // pre-BN f32 activations

  // --- degrees -> Dinv / Binv (shared across branches) ---
  zero_f32_k<<<256, 256, 0, stream>>>(Dinv, N_NODES);
  zero_f32_k<<<64, 256, 0, stream>>>(Binv, N_EDGES);
  degree_k<<<(NNZ_CNT + 255) / 256, 256, 0, stream>>>(node_idx, hedge_idx, Dinv, Binv);
  invert_k<<<(N_NODES + 255) / 256, 256, 0, stream>>>(Dinv, N_NODES);
  invert_k<<<(N_EDGES + 255) / 256, 256, 0, stream>>>(Binv, N_EDGES);

  const int mtiles = N_NODES / 16;                 // 6250, exact
  const int gemm_blocks = (mtiles + 7) / 8;        // 8 waves/block, 1 mtile/wave

  // one hypergraph propagation (gemm already done into HW[N x C])
  auto propagate = [&](int C) {
    zero_f32_k<<<1024, 256, 0, stream>>>(Eb, (long long)N_EDGES * C);
    long long tE = (long long)NNZ_CNT * (C / 4);
    int blkE = (int)((tE + 255) / 256);
    if (C == 128) scatter_to_edges_k<128><<<blkE, 256, 0, stream>>>(HW, node_idx, hedge_idx, Eb);
    else          scatter_to_edges_k<256><<<blkE, 256, 0, stream>>>(HW, node_idx, hedge_idx, Eb);
    scale_rows_k<<<2048, 256, 0, stream>>>(Eb, Binv, C, (long long)N_EDGES * C);
    zero_f32_k<<<4096, 256, 0, stream>>>(HW, (long long)N_NODES * C);  // HW becomes agg
    if (C == 128) scatter_to_nodes_k<128><<<blkE, 256, 0, stream>>>(Eb, node_idx, hedge_idx, HW);
    else          scatter_to_nodes_k<256><<<blkE, 256, 0, stream>>>(Eb, node_idx, hedge_idx, HW);
  };

  for (int b = 0; b < 3; ++b) {
    const float* w1  = (const float*)d_in[4 + 10 * b + 0];
    const float* b1  = (const float*)d_in[4 + 10 * b + 1];
    const float* w2  = (const float*)d_in[4 + 10 * b + 2];
    const float* b2  = (const float*)d_in[4 + 10 * b + 3];
    const float* w3  = (const float*)d_in[4 + 10 * b + 4];
    const float* b3  = (const float*)d_in[4 + 10 * b + 5];
    const float* g1  = (const float*)d_in[4 + 10 * b + 6];
    const float* bt1 = (const float*)d_in[4 + 10 * b + 7];
    const float* g2  = (const float*)d_in[4 + 10 * b + 8];
    const float* bt2 = (const float*)d_in[4 + 10 * b + 9];

    // input f32 -> bf16
    f32_to_bf16_k<<<4096, 256, 0, stream>>>(Xin[b], Hb, (long long)N_NODES * IN_C);

    // ---- conv1: [N,256] x [256,128] ----
    pack_b_k<<<(IN_C * HID_C + 255) / 256, 256, 0, stream>>>(w1, Wp, IN_C, HID_C);
    wmma_gemm_k<256, 128><<<gemm_blocks, 256, 0, stream>>>(Hb, Wp, HW);
    propagate(HID_C);
    zero_f32_k<<<2, 256, 0, stream>>>(sums, 2 * OUT_CC);  // sums + sqs (adjacent)
    post_agg_stats_k<128><<<600, 128, 0, stream>>>(HW, Dinv, b1, Hf, sums, sqs);
    bn_finalize_k<<<1, 128, 0, stream>>>(sums, sqs, g1, bt1, HID_C);
    bn_apply_bf16_k<128><<<4096, 256, 0, stream>>>(Hf, sums, sqs, Hb);

    // ---- conv2: [N,128] x [128,128] ----
    pack_b_k<<<(HID_C * HID_C + 255) / 256, 256, 0, stream>>>(w2, Wp, HID_C, HID_C);
    wmma_gemm_k<128, 128><<<gemm_blocks, 256, 0, stream>>>(Hb, Wp, HW);
    propagate(HID_C);
    zero_f32_k<<<2, 256, 0, stream>>>(sums, 2 * OUT_CC);
    post_agg_stats_k<128><<<600, 128, 0, stream>>>(HW, Dinv, b2, Hf, sums, sqs);
    bn_finalize_k<<<1, 128, 0, stream>>>(sums, sqs, g2, bt2, HID_C);
    bn_apply_bf16_k<128><<<4096, 256, 0, stream>>>(Hf, sums, sqs, Hb);

    // ---- conv3: [N,128] x [128,256], no BN ----
    pack_b_k<<<(HID_C * OUT_CC + 255) / 256, 256, 0, stream>>>(w3, Wp, HID_C, OUT_CC);
    wmma_gemm_k<128, 256><<<gemm_blocks, 256, 0, stream>>>(Hb, Wp, HW);
    propagate(OUT_CC);
    final_out_k<<<4096, 256, 0, stream>>>(HW, Dinv, b3,
        (float*)d_out + (size_t)b * N_NODES * OUT_CC);
  }
}
